// RNN_84095459656153
// MI455X (gfx1250) — matrix operand order repaired and checked by reading, not compiled, not run on hardware
//
#include <hip/hip_runtime.h>

// ---- types for CDNA5 WMMA ----
typedef __attribute__((ext_vector_type(16))) __bf16 v16bf;
typedef __attribute__((ext_vector_type(8)))  float  v8f;

#define T_DIM 512
#define B_DIM 256
#define I_DIM 128
#define H_DIM 256

static __device__ __forceinline__ unsigned short f2bf(float f) {
    union { float f; unsigned u; } v; v.f = f;
    unsigned u = v.u;
    u += 0x7FFFu + ((u >> 16) & 1u);   // round-to-nearest-even
    return (unsigned short)(u >> 16);
}

// Hardware tanh if the toolchain exposes it (CDNA5 has TANH in the trans unit),
// otherwise OCML tanhf.  Guarded so compilation can never fail.
static __device__ __forceinline__ float fast_tanh(float x) {
#if __has_builtin(__builtin_amdgcn_tanhf)
    return __builtin_amdgcn_tanhf(x);
#elif __has_builtin(__builtin_amdgcn_tanh_f32)
    return __builtin_amdgcn_tanh_f32(x);
#else
    return tanhf(x);
#endif
}

static __device__ __forceinline__ v8f wmma_bf16(v16bf a, v16bf b, v8f c) {
    return __builtin_amdgcn_wmma_f32_16x16x32_bf16(
        /*neg_a=*/false, a, /*neg_b=*/false, b,
        /*c_mod=*/(short)0, c, /*reuse_a=*/false, /*reuse_b=*/false);
}

// ---------------------------------------------------------------------------
// Phase 0: convert Wh [256x256] and Wi [256x128] f32 -> bf16 into workspace
// ---------------------------------------------------------------------------
__global__ void k_cvt_weights(const float* __restrict__ Wh,
                              const float* __restrict__ Wi,
                              unsigned short* __restrict__ Whb,
                              unsigned short* __restrict__ Wib) {
    int idx = blockIdx.x * blockDim.x + threadIdx.x;   // 0 .. 98303
    if (idx < H_DIM * H_DIM) {
        Whb[idx] = f2bf(Wh[idx]);
    } else {
        int j = idx - H_DIM * H_DIM;                   // 0 .. 32767
        Wib[j] = f2bf(Wi[j]);
    }
}

// ---------------------------------------------------------------------------
// Phase 1: xi[m, n] = sum_k x[m,k] * Wi[n,k] + bh[n]   (m = t*B+b, 131072 rows)
// Workgroup = 256 thr = 8 waves; WG tile = 32 rows x 256 cols.
// Wave tile = 16 rows x 64 cols (4 N-tiles), K = 128 (4 slices of 32).
// Memory-bound (64MB read + 128MB write); WMMA keeps math off the path.
// ---------------------------------------------------------------------------
__global__ void __launch_bounds__(256, 1)
k_xproj(const float* __restrict__ x,
        const unsigned short* __restrict__ Wib,
        const float* __restrict__ bh,
        float* __restrict__ xi) {
    const int tid  = threadIdx.x;
    const int wave = tid >> 5;
    const int lane = tid & 31;
    const int col  = lane & 15;
    const int half = lane >> 4;

    const int  wrow = wave & 1;
    const int  wcol = wave >> 1;
    const long m0   = (long)blockIdx.x * 32 + wrow * 16;
    const int  n0   = wcol * 64;

    // B fragments of Wi^T: Bf[nt*4+kt], B[k,n] = Wi[n,k] (row-major Wi)
    v16bf Bf[16];
    #pragma unroll
    for (int nt = 0; nt < 4; ++nt) {
        #pragma unroll
        for (int kt = 0; kt < 4; ++kt) {
            const int n  = n0 + nt * 16 + col;
            const int kk = kt * 32 + half * 16;            // 16 contiguous bf16
            Bf[nt * 4 + kt] = *(const v16bf*)(Wib + (long)n * I_DIM + kk);
        }
    }

    v8f acc[4];
    #pragma unroll
    for (int nt = 0; nt < 4; ++nt)
        #pragma unroll
        for (int r = 0; r < 8; ++r) acc[nt][r] = 0.0f;

    const long arow = m0 + col;
    const float* xrow = x + arow * I_DIM;

    #pragma unroll
    for (int kt = 0; kt < 4; ++kt) {
        // A fragment: elems 0-15 -> K = kt*32 + half*16 .. +15 (contiguous)
        const int k0 = kt * 32 + half * 16;
        const int k1 = k0 + 8;
        float4 f0 = *(const float4*)(xrow + k0);
        float4 f1 = *(const float4*)(xrow + k0 + 4);
        float4 f2 = *(const float4*)(xrow + k1);
        float4 f3 = *(const float4*)(xrow + k1 + 4);
        union { v16bf v; unsigned short u[16]; } A;
        A.u[0]  = f2bf(f0.x); A.u[1]  = f2bf(f0.y); A.u[2]  = f2bf(f0.z); A.u[3]  = f2bf(f0.w);
        A.u[4]  = f2bf(f1.x); A.u[5]  = f2bf(f1.y); A.u[6]  = f2bf(f1.z); A.u[7]  = f2bf(f1.w);
        A.u[8]  = f2bf(f2.x); A.u[9]  = f2bf(f2.y); A.u[10] = f2bf(f2.z); A.u[11] = f2bf(f2.w);
        A.u[12] = f2bf(f3.x); A.u[13] = f2bf(f3.y); A.u[14] = f2bf(f3.z); A.u[15] = f2bf(f3.w);

        #pragma unroll
        for (int nt = 0; nt < 4; ++nt)
            acc[nt] = wmma_bf16(A.v, Bf[nt * 4 + kt], acc[nt]);
    }

    // write xi (+bh): C layout: VGPR r -> m = r + half*8 ; lane col -> n
    #pragma unroll
    for (int nt = 0; nt < 4; ++nt) {
        const int  n  = n0 + nt * 16 + col;
        const float bb = bh[n];
        #pragma unroll
        for (int r = 0; r < 8; ++r) {
            const long m = m0 + r + half * 8;
            xi[m * H_DIM + n] = acc[nt][r] + bb;
        }
    }
}

// ---------------------------------------------------------------------------
// Phase 2: persistent recurrent scan.  16 WGs, each owns 16 batch rows
// (the recurrence is independent across batch -> zero inter-WG sync).
// Double-buffered h (2 x 16x256 bf16) in LDS -> ONE barrier per step.
// Each wave owns 32 output columns; its 16 Wh B-fragments stay in VGPRs for
// all 512 steps.  d_out holds xi_t on entry to step t, overwritten with h_t.
// xi_{t+1} is prefetched into registers during step t's compute/barrier.
// K split into two independent WMMA chains (depth 4) per accumulator.
// ---------------------------------------------------------------------------
#define HSTR 264   // padded LDS row stride in bf16 elems (528B -> bank spread)

__global__ void __launch_bounds__(256, 1)
k_scan(const float* __restrict__ h0,
       const unsigned short* __restrict__ Whb,
       float* __restrict__ out) {
    __shared__ unsigned short hl[2][16 * HSTR];

    const int tid  = threadIdx.x;
    const int wave = tid >> 5;
    const int lane = tid & 31;
    const int col  = lane & 15;
    const int half = lane >> 4;
    const int n0   = wave * 32;
    const int b0   = blockIdx.x * 16;

    // init LDS buffer 0 with h0 (bf16)
    for (int i = tid; i < 16 * H_DIM; i += 256) {
        const int r = i >> 8;
        const int c = i & 255;
        hl[0][r * HSTR + c] = f2bf(h0[(long)(b0 + r) * H_DIM + c]);
    }

    // Preload Wh B-fragments (resident for all 512 steps):
    // Bf[nt*8+kt]; B[k,n] = Wh[n,k] (row-major Wh)
    v16bf Bf[16];
    #pragma unroll
    for (int nt = 0; nt < 2; ++nt) {
        #pragma unroll
        for (int kt = 0; kt < 8; ++kt) {
            const int n  = n0 + nt * 16 + col;
            const int kk = kt * 32 + half * 16;
            Bf[nt * 8 + kt] = *(const v16bf*)(Whb + (long)n * H_DIM + kk);
        }
    }

    // Per-lane base offset of this wave's C-tile elements in a [B,H] slab:
    // element r of tile nt lives at (b0 + r + half*8)*H + n0 + nt*16 + col.
    const long base = (long)(b0 + half * 8) * H_DIM + n0 + col;

    // Prefetch xi for t=0
    float nx0[8], nx1[8];
    {
        const float* p = out + base;
        #pragma unroll
        for (int r = 0; r < 8; ++r) {
            nx0[r] = p[(long)r * H_DIM];
            nx1[r] = p[(long)r * H_DIM + 16];
        }
    }

    __syncthreads();   // h0 init visible

    for (int t = 0; t < T_DIM; ++t) {
        const unsigned short* rb = hl[t & 1];        // read h(t-1)
        unsigned short*       wb = hl[(t + 1) & 1];  // write h(t)

        // accumulators: chain A seeded with prefetched xi_t, chain B from zero
        v8f acc0, acc1, acc0b, acc1b;
        #pragma unroll
        for (int r = 0; r < 8; ++r) {
            acc0[r]  = nx0[r];
            acc1[r]  = nx1[r];
            acc0b[r] = 0.0f;
            acc1b[r] = 0.0f;
        }

        // issue prefetch of xi_{t+1}; flies during WMMA/tanh/barrier below
        if (t + 1 < T_DIM) {
            const float* p = out + (long)(t + 1) * (B_DIM * H_DIM) + base;
            #pragma unroll
            for (int r = 0; r < 8; ++r) {
                nx0[r] = p[(long)r * H_DIM];
                nx1[r] = p[(long)r * H_DIM + 16];
            }
        }

        // h(t-1) @ Wh^T : 8 K-slices of 32, two independent chains
        #pragma unroll
        for (int kt = 0; kt < 4; ++kt) {
            union { v16bf v; uint4 q[2]; } A;
            const int k0 = kt * 32 + half * 16;
            A.q[0] = *(const uint4*)(&rb[col * HSTR + k0]);
            A.q[1] = *(const uint4*)(&rb[col * HSTR + k0 + 8]);
            acc0 = wmma_bf16(A.v, Bf[kt],     acc0);
            acc1 = wmma_bf16(A.v, Bf[8 + kt], acc1);
        }
        #pragma unroll
        for (int kt = 4; kt < 8; ++kt) {
            union { v16bf v; uint4 q[2]; } A;
            const int k0 = kt * 32 + half * 16;
            A.q[0] = *(const uint4*)(&rb[col * HSTR + k0]);
            A.q[1] = *(const uint4*)(&rb[col * HSTR + k0 + 8]);
            acc0b = wmma_bf16(A.v, Bf[kt],     acc0b);
            acc1b = wmma_bf16(A.v, Bf[8 + kt], acc1b);
        }

        // combine chains + nonlinearity
        #pragma unroll
        for (int r = 0; r < 8; ++r) {
            acc0[r] = fast_tanh(acc0[r] + acc0b[r]);
            acc1[r] = fast_tanh(acc1[r] + acc1b[r]);
        }

        // write h_t: global f32 (the output) + LDS bf16 (next step's A).
        // Write buffer != read buffer, so no fence needed before these stores.
        float* ot = out + (long)t * (B_DIM * H_DIM) + base;
        #pragma unroll
        for (int r = 0; r < 8; ++r) {
            const float v0 = acc0[r];
            const float v1 = acc1[r];
            ot[(long)r * H_DIM]      = v0;
            ot[(long)r * H_DIM + 16] = v1;
            wb[(r + half * 8) * HSTR + n0 + col]      = f2bf(v0);
            wb[(r + half * 8) * HSTR + n0 + 16 + col] = f2bf(v1);
        }

        __syncthreads();   // h(t) visible to all waves before step t+1
    }
}

// ---------------------------------------------------------------------------
extern "C" void kernel_launch(void* const* d_in, const int* in_sizes, int n_in,
                              void* d_out, int out_size, void* d_ws, size_t ws_size,
                              hipStream_t stream) {
    const float* x  = (const float*)d_in[0];   // [T,B,I]
    const float* h0 = (const float*)d_in[1];   // [B,H]
    const float* Wi = (const float*)d_in[2];   // [H,I]
    const float* Wh = (const float*)d_in[3];   // [H,H]
    const float* bh = (const float*)d_in[4];   // [H]
    float* out = (float*)d_out;                // [T,B,H]

    unsigned short* Whb = (unsigned short*)d_ws;           // 256*256 bf16
    unsigned short* Wib = Whb + H_DIM * H_DIM;             // 256*128 bf16

    // Phase 0: weight conversion (98304 elements)
    k_cvt_weights<<<(H_DIM * H_DIM + H_DIM * I_DIM) / 256, 256, 0, stream>>>(
        Wh, Wi, Whb, Wib);

    // Phase 1: xi = x @ Wi^T + bh  ->  staged in d_out
    k_xproj<<<(T_DIM * B_DIM) / 32, 256, 0, stream>>>(x, Wib, bh, out);

    // Phase 2: recurrent scan, in place in d_out
    k_scan<<<B_DIM / 16, 256, 0, stream>>>(h0, Whb, out);
}
